// TriangleCrossAttention_65876208386294
// MI455X (gfx1250) — matrix operand order repaired and checked by reading, not compiled, not run on hardware
//
#include <hip/hip_runtime.h>

typedef __attribute__((ext_vector_type(16))) _Float16 v16h;
typedef __attribute__((ext_vector_type(8)))  float    v8f;
typedef __attribute__((ext_vector_type(4)))  float    f32x4;
typedef __attribute__((ext_vector_type(4)))  unsigned u32x4;
typedef __attribute__((ext_vector_type(2)))  unsigned u32x2;

#define IDIM 512
#define JDIM 512
#define KDIM 512
#define CQn  64
#define JBLK 64
#define L2E  1.4426950408889634f

// ---- LDS layout (bytes). Strides padded for bank-conflict-free access. ----
#define OFF_K    0        // sK  : 512 x 72 f16  (k row-major, padded)      73728
#define OFF_VT   73728    // sVT : 64 x 520 f16  (v transposed, padded)     66560
#define OFF_XN   140288   // sXN : 64 x 72 f16   (layernormed x)             9216
#define OFF_WQT  149504   // sWqT: 64 x 72 f16   WqT[hc][cq]                 9216
#define OFF_WGT  158720   // sWgT: 64 x 72 f16                               9216
#define OFF_WOT  167936   // sWoT: 64 x 72 f16   WoT[cq][hc]                 9216
#define OFF_WKF  177152   // sWkF: 8 x 64 f32                                2048
#define OFF_WVF  179200   // sWvF: 8 x 64 f32                                2048
#define OFF_BG   181248   // sBG : 64 f32                                     256
#define OFF_BO   181504   // sBO : 64 f32                                     256
#define OFF_OG   181760   // sOG : 64 x 72 f16   gated o, [j][hc]            9216
#define OFF_OUT  190976   // sOut: 64 x 68 f32   out staging (17408)
#define OFF_Z    181760   // sZ  : 512 x 24 f16 (halves 8..15 zeroed), 24576
                          //       aliases sOG+sOut (disjoint lifetimes)
#define LDS_BYTES 208384

__device__ __forceinline__ unsigned pkh(float a, float b) {
  auto r = __builtin_amdgcn_cvt_pkrtz(a, b);   // v_cvt_pk_f16_f32 (__fp16 x2)
  return __builtin_bit_cast(unsigned, r);
}

union Frag { v16h h; u32x4 q[2]; };

// Build q B-fragment (scaled, D->B layout swap) and sigmoid gate for one j-tile.
__device__ __forceinline__ void qg_tiles(const _Float16* sXN, const _Float16* sWqT,
                                         const _Float16* sWgT, const float* sBG,
                                         int h, int jl, int l16, int lhalf,
                                         Frag& qB, float gt[8])
{
  const u32x4 zq = {0u, 0u, 0u, 0u};
  v8f qD = {}; v8f gD = {};
  #pragma unroll
  for (int ck = 0; ck < 2; ++ck) {
    Frag bx, aq, ag;
    bx.q[0] = *(const u32x4*)&sXN[jl * 72 + ck * 32 + lhalf * 8];
    bx.q[1] = *(const u32x4*)&sXN[jl * 72 + ck * 32 + 16 + lhalf * 8];
    int arow = (h * 16 + l16) * 72;
    aq.q[0] = *(const u32x4*)&sWqT[arow + ck * 32 + lhalf * 8];
    aq.q[1] = *(const u32x4*)&sWqT[arow + ck * 32 + 16 + lhalf * 8];
    ag.q[0] = *(const u32x4*)&sWgT[arow + ck * 32 + lhalf * 8];
    ag.q[1] = *(const u32x4*)&sWgT[arow + ck * 32 + 16 + lhalf * 8];
    qD = __builtin_amdgcn_wmma_f32_16x16x32_f16(false, aq.h, false, bx.h, (short)0, qD, false, false);
    gD = __builtin_amdgcn_wmma_f32_16x16x32_f16(false, ag.h, false, bx.h, (short)0, gD, false, false);
  }
  #pragma unroll
  for (int r = 0; r < 8; ++r) qD[r] *= 0.25f;        // 1/sqrt(CH)
  u32x4 o = { pkh(qD[0],qD[1]), pkh(qD[2],qD[3]), pkh(qD[4],qD[5]), pkh(qD[6],qD[7]) };
  u32x4 rv;
  rv.x = (unsigned)__shfl_xor((int)o.x, 16, 32);
  rv.y = (unsigned)__shfl_xor((int)o.y, 16, 32);
  rv.z = (unsigned)__shfl_xor((int)o.z, 16, 32);
  rv.w = (unsigned)__shfl_xor((int)o.w, 16, 32);
  qB.q[0] = o;                                       // lane half h holds K=8h..8h+7
  qB.q[1] = zq;                                      // K=16..31 -> zero
  #pragma unroll
  for (int r = 0; r < 8; ++r) {
    float v = gD[r] + sBG[h * 16 + lhalf * 8 + r];
    gt[r] = 1.f / (1.f + exp2f(-v * L2E));
  }
}

// One online-softmax step for a 32-key chunk: updates m/l, rescales acc,
// and produces the B-layout probability fragment (packed cross-half exchange).
__device__ __forceinline__ void soft_step(const v8f& s0, const v8f& s1, int lhalf,
                                          float& m_run, float& l_run, v8f& acc, Frag& pB)
{
  float cm = s0[0];
  #pragma unroll
  for (int r = 1; r < 8; ++r) cm = fmaxf(cm, s0[r]);
  #pragma unroll
  for (int r = 0; r < 8; ++r) cm = fmaxf(cm, s1[r]);
  cm = fmaxf(cm, __shfl_xor(cm, 16, 32));
  float m_new = fmaxf(m_run, cm);
  float so = exp2f((m_run - m_new) * L2E);
  float p[16]; float ps = 0.f;
  #pragma unroll
  for (int r = 0; r < 8; ++r) { p[r]     = exp2f((s0[r] - m_new) * L2E); ps += p[r]; }
  #pragma unroll
  for (int r = 0; r < 8; ++r) { p[8 + r] = exp2f((s1[r] - m_new) * L2E); ps += p[8 + r]; }
  ps += __shfl_xor(ps, 16, 32);
  l_run = l_run * so + ps;
  m_run = m_new;
  #pragma unroll
  for (int r = 0; r < 8; ++r) acc[r] *= so;
  u32x4 w0 = { pkh(p[0],p[1]),  pkh(p[2],p[3]),  pkh(p[4],p[5]),  pkh(p[6],p[7])  };
  u32x4 w1 = { pkh(p[8],p[9]),  pkh(p[10],p[11]),pkh(p[12],p[13]),pkh(p[14],p[15])};
  u32x4 sel = lhalf ? w0 : w1;     // send the half the partner needs
  u32x4 rv;
  rv.x = (unsigned)__shfl_xor((int)sel.x, 16, 32);
  rv.y = (unsigned)__shfl_xor((int)sel.y, 16, 32);
  rv.z = (unsigned)__shfl_xor((int)sel.z, 16, 32);
  rv.w = (unsigned)__shfl_xor((int)sel.w, 16, 32);
  pB.q[0] = lhalf ? rv : w0;
  pB.q[1] = lhalf ? w1 : rv;
}

__global__ __launch_bounds__(256, 1)
void tca_kernel(const float* __restrict__ x,   const float* __restrict__ z,
                const float* __restrict__ mask,const float* __restrict__ lnw,
                const float* __restrict__ lnb, const float* __restrict__ Wq,
                const float* __restrict__ Wk,  const float* __restrict__ Wv,
                const float* __restrict__ Wg,  const float* __restrict__ bg,
                const float* __restrict__ Wo,  const float* __restrict__ bo,
                float* __restrict__ out)
{
  extern __shared__ char smem[];
  _Float16* sK   = (_Float16*)(smem + OFF_K);
  _Float16* sVT  = (_Float16*)(smem + OFF_VT);
  _Float16* sXN  = (_Float16*)(smem + OFF_XN);
  _Float16* sWqT = (_Float16*)(smem + OFF_WQT);
  _Float16* sWgT = (_Float16*)(smem + OFF_WGT);
  _Float16* sWoT = (_Float16*)(smem + OFF_WOT);
  float*    sWkF = (float*)   (smem + OFF_WKF);
  float*    sWvF = (float*)   (smem + OFF_WVF);
  float*    sBG  = (float*)   (smem + OFF_BG);
  float*    sBO  = (float*)   (smem + OFF_BO);
  _Float16* sOG  = (_Float16*)(smem + OFF_OG);
  float*    sOut = (float*)   (smem + OFF_OUT);
  _Float16* sZ   = (_Float16*)(smem + OFF_Z);    // aliases sOG/sOut (phase 2 only)

  const int t  = threadIdx.x;
  const int i  = blockIdx.x;
  const int jb = blockIdx.y * JBLK;
  const int lane = t & 31;
  const int l16 = lane & 15, lhalf = lane >> 4;
  // wave index as provable SGPR so wave-level branches stay scalar
  const int wq = __builtin_amdgcn_readfirstlane(t >> 5);
  const u32x4 zq = {0u, 0u, 0u, 0u};

  // NOTE on mask: the reference's mask bias indexes the *query* axis and
  // broadcasts over keys (uniform along the softmax axis). Softmax is exactly
  // shift-invariant, so the bias cannot change the output; omitted.
  (void)mask;

  // ---------------- Phase 1: weights -> LDS ----------------
  {
    int r = t >> 2, cg = (t & 3) * 16;
    const float* srcs[3] = {Wq, Wg, Wo};
    _Float16*    dsts[3] = {sWqT, sWgT, sWoT};
    #pragma unroll
    for (int m = 0; m < 3; ++m) {
      const float* s = srcs[m] + r * 64 + cg;
      _Float16*    d = dsts[m];
      #pragma unroll
      for (int q = 0; q < 4; ++q) {
        f32x4 v = *(const f32x4*)(s + q * 4);
        d[(cg + q*4 + 0) * 72 + r] = (_Float16)v.x;   // transpose scatter
        d[(cg + q*4 + 1) * 72 + r] = (_Float16)v.y;
        d[(cg + q*4 + 2) * 72 + r] = (_Float16)v.z;
        d[(cg + q*4 + 3) * 72 + r] = (_Float16)v.w;
      }
    }
    if (t < 128) { *(f32x4*)(sWkF + t * 4)        = *(const f32x4*)(Wk + t * 4); }
    else         { *(f32x4*)(sWvF + (t-128) * 4)  = *(const f32x4*)(Wv + (t-128) * 4); }
    if (t < 64)  { sBG[t] = bg[t]; sBO[t] = bo[t]; }
  }
  __syncthreads();

  // ------- Phase 2a: stage z as f16 (upper half pre-zeroed) + LayerNorm ----
  {
    #pragma unroll
    for (int kk = 0; kk < 2; ++kk) {
      int key = t * 2 + kk;
      const float* zp = z + ((size_t)i * KDIM + key) * 8;
      f32x4 z0 = *(const f32x4*)zp;
      f32x4 z1 = *(const f32x4*)(zp + 4);
      u32x4 zp16 = { pkh(z0.x, z0.y), pkh(z0.z, z0.w),
                     pkh(z1.x, z1.y), pkh(z1.z, z1.w) };
      *(u32x4*)&sZ[key * 24]     = zp16;
      *(u32x4*)&sZ[key * 24 + 8] = zq;               // c=8..15 pad = 0
    }
  }
  if (t < 64) {
    int j = t;
    const float* xp = x + ((size_t)i * JDIM + jb + j) * CQn;
    float s = 0.f, s2 = 0.f;
    #pragma unroll
    for (int q = 0; q < 16; ++q) {
      f32x4 v = *(const f32x4*)(xp + q * 4);
      s  += v.x + v.y + v.z + v.w;
      s2 += v.x*v.x + v.y*v.y + v.z*v.z + v.w*v.w;
    }
    float mu  = s * (1.f / 64.f);
    float var = s2 * (1.f / 64.f) - mu * mu;
    float rs  = rsqrtf(var + 1e-5f);
    #pragma unroll
    for (int q = 0; q < 16; ++q) {
      f32x4 v = *(const f32x4*)(xp + q * 4);
      float n0 = (v.x - mu) * rs * lnw[q*4+0] + lnb[q*4+0];
      float n1 = (v.y - mu) * rs * lnw[q*4+1] + lnb[q*4+1];
      float n2 = (v.z - mu) * rs * lnw[q*4+2] + lnb[q*4+2];
      float n3 = (v.w - mu) * rs * lnw[q*4+3] + lnb[q*4+3];
      u32x2 pp = { pkh(n0, n1), pkh(n2, n3) };
      *(u32x2*)&sXN[j * 72 + q * 4] = pp;
    }
  }
  __syncthreads();

  // ---------------- Phase 2b: k/v projection via WMMA ----------------------
  // W fragment bytes are identical as A-operand (k path) or B-operand (v path).
  {
    int mat    = wq & 1;                // 0 -> k, 1 -> v (scalar branch)
    int ktbase = (wq >> 1) * 8;
    const float* W = mat ? sWvF : sWkF;
    Frag bw[4];
    #pragma unroll
    for (int ht = 0; ht < 4; ++ht) {
      int hc = ht * 16 + l16;
      u32x4 bb = { pkh(W[0*64+hc], W[1*64+hc]), pkh(W[2*64+hc], W[3*64+hc]),
                   pkh(W[4*64+hc], W[5*64+hc]), pkh(W[6*64+hc], W[7*64+hc]) };
      bw[ht].q[0] = lhalf ? zq : bb;    // rows c=8..15 & lanes>=16: zero pad
      bw[ht].q[1] = zq;
    }
    if (mat == 0) {
      // k^T = WkT(A) x z^T(B): D lane = key, VGPRs = 8 consecutive hc
      // -> one b128 store per tile into row-major sK[key][hc].
      for (int n = 0; n < 8; ++n) {
        int kt = ktbase + n;
        Frag bz;
        bz.q[0] = *(const u32x4*)&sZ[(kt * 16 + l16) * 24]; // rows c=0..7 (real)
        bz.q[1] = zq;                                       // rows c=8..15 -> 0
        #pragma unroll
        for (int ht = 0; ht < 4; ++ht) {
          v8f zc = {};
          v8f d = __builtin_amdgcn_wmma_f32_16x16x32_f16(false, bw[ht].h, false, bz.h, (short)0, zc, false, false);
          u32x4 vv = { pkh(d[0],d[1]), pkh(d[2],d[3]), pkh(d[4],d[5]), pkh(d[6],d[7]) };
          *(u32x4*)&sK[(kt * 16 + l16) * 72 + ht * 16 + 8 * lhalf] = vv;
        }
      }
    } else {
      // v = z(A) x Wv(B): D lane = hc, VGPRs = 8 consecutive keys
      // -> one b128 store per tile into transposed sVT[hc][key].
      for (int n = 0; n < 8; ++n) {
        int kt = ktbase + n;
        Frag az;
        az.q[0] = *(const u32x4*)&sZ[(kt * 16 + l16) * 24 + lhalf * 8]; // hi lanes read zeros
        az.q[1] = zq;
        #pragma unroll
        for (int ht = 0; ht < 4; ++ht) {
          v8f zc = {};
          v8f d = __builtin_amdgcn_wmma_f32_16x16x32_f16(false, az.h, false, bw[ht].h, (short)0, zc, false, false);
          u32x4 vv = { pkh(d[0],d[1]), pkh(d[2],d[3]), pkh(d[4],d[5]), pkh(d[6],d[7]) };
          *(u32x4*)&sVT[(ht * 16 + l16) * 520 + kt * 16 + 8 * lhalf] = vv;
        }
      }
    }
  }
  __syncthreads();

  // ------- Phase 3: attention, two j-tiles of the SAME head per wave -------
  // Waves 2h and 2h+1 cover head h; shared sK/sVT fragments are loaded once
  // and the two independent softmax streams interleave to hide WMMA hazards.
  {
    int h   = wq >> 1;
    int jg  = (wq & 1) * 2;
    int jl0 = (jg + 0) * 16 + l16;
    int jl1 = (jg + 1) * 16 + l16;

    Frag qB0, qB1; float gt0[8], gt1[8];
    qg_tiles(sXN, sWqT, sWgT, sBG, h, jl0, l16, lhalf, qB0, gt0);
    qg_tiles(sXN, sWqT, sWgT, sBG, h, jl1, l16, lhalf, qB1, gt1);

    float m0 = -3.0e38f, l0 = 0.f, m1 = -3.0e38f, l1 = 0.f;
    v8f acc0 = {}, acc1 = {};

    for (int ch = 0; ch < 16; ++ch) {
      int kb = ch * 32;
      Frag a0, a1;
      a0.q[0] = *(const u32x4*)&sK[(kb + l16) * 72 + h * 16 + lhalf * 8];      a0.q[1] = zq;
      a1.q[0] = *(const u32x4*)&sK[(kb + 16 + l16) * 72 + h * 16 + lhalf * 8]; a1.q[1] = zq;
      Frag av;
      int vrow = (h * 16 + l16) * 520;
      av.q[0] = *(const u32x4*)&sVT[vrow + kb + lhalf * 16];
      av.q[1] = *(const u32x4*)&sVT[vrow + kb + lhalf * 16 + 8];

      v8f zc = {};
      v8f s0a = __builtin_amdgcn_wmma_f32_16x16x32_f16(false, a0.h, false, qB0.h, (short)0, zc, false, false);
      v8f s1a = __builtin_amdgcn_wmma_f32_16x16x32_f16(false, a1.h, false, qB0.h, (short)0, zc, false, false);
      v8f s0b = __builtin_amdgcn_wmma_f32_16x16x32_f16(false, a0.h, false, qB1.h, (short)0, zc, false, false);
      v8f s1b = __builtin_amdgcn_wmma_f32_16x16x32_f16(false, a1.h, false, qB1.h, (short)0, zc, false, false);

      Frag pBa, pBb;
      soft_step(s0a, s1a, lhalf, m0, l0, acc0, pBa);
      soft_step(s0b, s1b, lhalf, m1, l1, acc1, pBb);

      acc0 = __builtin_amdgcn_wmma_f32_16x16x32_f16(false, av.h, false, pBa.h, (short)0, acc0, false, false);
      acc1 = __builtin_amdgcn_wmma_f32_16x16x32_f16(false, av.h, false, pBb.h, (short)0, acc1, false, false);
    }

    float i0 = 1.f / l0, i1 = 1.f / l1;
    float ov0[8], ov1[8];
    #pragma unroll
    for (int r = 0; r < 8; ++r) { ov0[r] = acc0[r] * i0 * gt0[r]; ov1[r] = acc1[r] * i1 * gt1[r]; }
    u32x4 og0 = { pkh(ov0[0],ov0[1]), pkh(ov0[2],ov0[3]), pkh(ov0[4],ov0[5]), pkh(ov0[6],ov0[7]) };
    u32x4 og1 = { pkh(ov1[0],ov1[1]), pkh(ov1[2],ov1[3]), pkh(ov1[4],ov1[5]), pkh(ov1[6],ov1[7]) };
    *(u32x4*)&sOG[jl0 * 72 + h * 16 + lhalf * 8] = og0;
    *(u32x4*)&sOG[jl1 * 72 + h * 16 + lhalf * 8] = og1;
  }
  __syncthreads();

  // ---------------- Phase 4: out^T = WoT x OG^T + bo ----------------
  for (int idx = 0; idx < 2; ++idx) {
    int tile = wq * 2 + idx;
    int ct = tile & 3, jt2 = tile >> 2;
    int jl = jt2 * 16 + l16;
    v8f od = {};
    #pragma unroll
    for (int ck = 0; ck < 2; ++ck) {
      Frag aw, bb;
      int arow = (ct * 16 + l16) * 72;
      aw.q[0] = *(const u32x4*)&sWoT[arow + ck * 32 + lhalf * 8];
      aw.q[1] = *(const u32x4*)&sWoT[arow + ck * 32 + 16 + lhalf * 8];
      bb.q[0] = *(const u32x4*)&sOG[jl * 72 + ck * 32 + lhalf * 8];
      bb.q[1] = *(const u32x4*)&sOG[jl * 72 + ck * 32 + 16 + lhalf * 8];
      od = __builtin_amdgcn_wmma_f32_16x16x32_f16(false, aw.h, false, bb.h, (short)0, od, false, false);
    }
    #pragma unroll
    for (int r = 0; r < 8; ++r) od[r] += sBO[ct * 16 + lhalf * 8 + r];
    f32x4 o0 = { od[0], od[1], od[2], od[3] };
    f32x4 o1 = { od[4], od[5], od[6], od[7] };
    *(f32x4*)&sOut[jl * 68 + ct * 16 + lhalf * 8]     = o0;
    *(f32x4*)&sOut[jl * 68 + ct * 16 + lhalf * 8 + 4] = o1;
  }
  __syncthreads();

  // ---------------- Phase 5: coalesced store ----------------
  {
    int row = t >> 2, cg = (t & 3) * 16;
    float* op = out + ((size_t)i * JDIM + jb + row) * CQn + cg;
    #pragma unroll
    for (int q = 0; q < 4; ++q) {
      f32x4 v = *(const f32x4*)&sOut[row * 68 + cg + q * 4];
      *(f32x4*)(op + q * 4) = v;
    }
  }
}

extern "C" void kernel_launch(void* const* d_in, const int* in_sizes, int n_in,
                              void* d_out, int out_size, void* d_ws, size_t ws_size,
                              hipStream_t stream) {
  (void)in_sizes; (void)n_in; (void)out_size; (void)d_ws; (void)ws_size;
  const float* x    = (const float*)d_in[0];
  const float* z    = (const float*)d_in[1];
  const float* mask = (const float*)d_in[2];
  const float* lnw  = (const float*)d_in[3];
  const float* lnb  = (const float*)d_in[4];
  const float* Wq   = (const float*)d_in[5];
  const float* Wk   = (const float*)d_in[6];
  const float* Wv   = (const float*)d_in[7];
  const float* Wg   = (const float*)d_in[8];
  const float* bg   = (const float*)d_in[9];
  const float* Wo   = (const float*)d_in[10];
  const float* bo   = (const float*)d_in[11];

  (void)hipFuncSetAttribute((const void*)tca_kernel,
                            hipFuncAttributeMaxDynamicSharedMemorySize, LDS_BYTES);

  dim3 grid(IDIM, JDIM / JBLK);
  tca_kernel<<<grid, dim3(256), LDS_BYTES, stream>>>(
      x, z, mask, lnw, lnb, Wq, Wk, Wv, Wg, bg, Wo, bo, (float*)d_out);
}
